// GRAFTNET_67053029425241
// MI455X (gfx1250) — matrix operand order repaired and checked by reading, not compiled, not run on hardware
//
#include <hip/hip_runtime.h>
#include <math.h>

// ---------------- constants (match reference) ----------------
#define Bc 8
#define Ec 2000
#define Fc 6000
#define Qc 20
#define Dc 256
#define Lc 3
#define NNZc (2 * Bc * Fc)      // 96000
#define NUM_ENTITYc 400000
#define NUM_WORDc 60000
#define PR_LAMBDA 0.8f
#define FACT_SCALEc 3.0f
#define VERY_NEGc -100000000000.0f
#define VERY_SMALLc 1e-10f

typedef __attribute__((ext_vector_type(16))) __bf16 v16bf;
typedef __attribute__((ext_vector_type(8)))  __bf16 v8bf;
typedef __attribute__((ext_vector_type(8)))  float  v8f;
typedef __attribute__((ext_vector_type(4)))  float  v4f;

__device__ __forceinline__ float sigmoidf_(float x) { return 1.0f / (1.0f + expf(-x)); }

// ======================================================================
// WMMA GEMM:  C[M,N] = act( alpha * A[M,K] @ W[N,K]^T + bias[N] )
// A rows optionally gathered:  A_row(r) = A + gidx[r]*lda (embedding tables)
// Block tile 64x64, 8 waves; each wave owns two 16x16 WMMA subtiles
// (shared A fragment). Double-buffered LDS, one barrier per K-step.
// Requires K % 32 == 0 (true for all call sites: K in {256, 768}).
// All A/W call sites are 16B aligned with lda/ldw % 8 == 0.
// ======================================================================
#define BM 64
#define BNT 64
#define BK 32
#define LDP 40   // padded LDS row (bf16): 80B stride, multiple of 16B

template<bool GATHER>
__global__ __launch_bounds__(256) void k_gemm_bf16(
    const float* __restrict__ A, int lda, const int* __restrict__ gidx,
    const float* __restrict__ W, int ldw,
    const float* __restrict__ bias,
    float* __restrict__ C, int ldc,
    int M, int N, int K, float alpha, int relu)
{
    __shared__ __align__(16) __bf16 As[2][BM][LDP];
    __shared__ __align__(16) __bf16 Ws[2][BNT][LDP];

    const int m0   = blockIdx.x * BM;
    const int n0   = blockIdx.y * BNT;
    const int tid  = threadIdx.x;
    const int wave = tid >> 5;
    const int lane = tid & 31;
    const int mw   = (wave & 3) * 16;          // wave's M subtile
    const int nb   = (wave >> 2) * 32;         // wave's N pair base (0 or 32)
    const int half = lane >> 4;
    const int ml   = lane & 15;

    // ---- per-thread staging coordinates (same for A and W tiles) ----
    const int sr = tid >> 2;                   // 0..63 (tile row)
    const int sc = (tid & 3) * 8;              // 0,8,16,24 (k offset)

    // branchless row setup: clamp OOB rows, zero via multiplier
    const int  agr = m0 + sr;
    const int  agrc = agr < M ? agr : (M - 1);
    const float azm = agr < M ? 1.0f : 0.0f;
    const long arb = GATHER ? (long)gidx[agrc] : (long)agrc;
    const float* arow = A + arb * (long)lda + sc;

    const int  wgr = n0 + sr;
    const int  wgrc = wgr < N ? wgr : (N - 1);
    const float wzm = wgr < N ? 1.0f : 0.0f;
    const float* wrow = W + (long)wgrc * ldw + sc;

    auto stage = [&](int kt, int buf) {
        const int k0 = kt * BK;
        {   // A tile: two v4f loads -> v8bf packed LDS store (16B)
            v4f x0 = *(const v4f*)(arow + k0);
            v4f x1 = *(const v4f*)(arow + k0 + 4);
            x0 *= azm; x1 *= azm;
            v8bf pk;
#pragma unroll
            for (int j = 0; j < 4; ++j) { pk[j] = (__bf16)x0[j]; pk[4 + j] = (__bf16)x1[j]; }
            *(v8bf*)&As[buf][sr][sc] = pk;
        }
        {   // W tile
            v4f x0 = *(const v4f*)(wrow + k0);
            v4f x1 = *(const v4f*)(wrow + k0 + 4);
            x0 *= wzm; x1 *= wzm;
            v8bf pk;
#pragma unroll
            for (int j = 0; j < 4; ++j) { pk[j] = (__bf16)x0[j]; pk[4 + j] = (__bf16)x1[j]; }
            *(v8bf*)&Ws[buf][sr][sc] = pk;
        }
    };

    v8f acc0 = {};
    v8f acc1 = {};

    const int nk = K / BK;
    stage(0, 0);
    __syncthreads();

    for (int t = 0; t < nk; ++t) {
        const int cur = t & 1;
        if (t + 1 < nk) stage(t + 1, cur ^ 1);   // fill other buffer, no sync needed

        // ---- fragments: contiguous k = half*16 + s for both A and B ----
        v16bf af, bf0, bf1;
#pragma unroll
        for (int j = 0; j < 8; ++j) {
            af[j]     = As[cur][mw + ml][half * 16 + j];       // K = half*16 + j
            af[8 + j] = As[cur][mw + ml][half * 16 + 8 + j];   // K = half*16 + 8 + j
        }
#pragma unroll
        for (int j = 0; j < 16; ++j) {
            bf0[j] = Ws[cur][nb + ml][half * 16 + j];          // K = half*16 + j
            bf1[j] = Ws[cur][nb + 16 + ml][half * 16 + j];
        }
        acc0 = __builtin_amdgcn_wmma_f32_16x16x32_bf16(false, af, false, bf0, (short)0, acc0, false, false);
        acc1 = __builtin_amdgcn_wmma_f32_16x16x32_bf16(false, af, false, bf1, (short)0, acc1, false, false);
        __syncthreads();
    }

    // ---- epilogue: C vgpr r -> M = r + 8*half ----
#pragma unroll
    for (int s = 0; s < 2; ++s) {
        const int n = n0 + nb + s * 16 + ml;
        if (n < N) {
            const float bv = bias ? bias[n] : 0.0f;
            const v8f& a = s ? acc1 : acc0;
#pragma unroll
            for (int r = 0; r < 8; ++r) {
                const int m = m0 + mw + half * 8 + r;
                if (m < M) {
                    float v = a[r] * alpha + bv;
                    if (relu) v = fmaxf(v, 0.0f);
                    C[(long)m * ldc + n] = v;
                }
            }
        }
    }
}

// ======================================================================
// LSTM: one workgroup, 20 sequential steps; h/c/x/g staged in LDS.
// ======================================================================
__global__ __launch_bounds__(1024) void k_lstm(
    const int* __restrict__ qtext, const float* __restrict__ wtab,
    const float* __restrict__ Wih, const float* __restrict__ Whh,
    const float* __restrict__ bih, const float* __restrict__ bhh,
    float* __restrict__ qh)
{
    __shared__ float xs[Bc][Dc];
    __shared__ float hs[Bc][Dc];
    __shared__ float cs[Bc][Dc];
    __shared__ float gs[Bc][4 * Dc];
    const int tid = threadIdx.x;

    for (int i = tid; i < Bc * Dc; i += 1024) { hs[i >> 8][i & 255] = 0.0f; cs[i >> 8][i & 255] = 0.0f; }
    __syncthreads();

    for (int t = 0; t < Qc; ++t) {
        for (int i = tid; i < Bc * Dc; i += 1024) {
            int b = i >> 8, d = i & 255;
            int w = qtext[b * Qc + t];
            xs[b][d] = wtab[(long)w * Dc + d];
        }
        __syncthreads();

        {   // g[b][n] for n = tid
            const int n = tid;
            const float bb = bih[n] + bhh[n];
            const float* wi = Wih + (long)n * Dc;
            const float* wh = Whh + (long)n * Dc;
            for (int b = 0; b < Bc; ++b) {
                float acc = bb;
                for (int k = 0; k < Dc; ++k)
                    acc += xs[b][k] * wi[k] + hs[b][k] * wh[k];
                gs[b][n] = acc;
            }
        }
        __syncthreads();

        for (int i = tid; i < Bc * Dc; i += 1024) {
            int b = i >> 8, j = i & 255;
            float ig = sigmoidf_(gs[b][j]);
            float fg = sigmoidf_(gs[b][Dc + j]);
            float gg = tanhf(gs[b][2 * Dc + j]);
            float og = sigmoidf_(gs[b][3 * Dc + j]);
            float c = fg * cs[b][j] + ig * gg;
            float h = og * tanhf(c);
            cs[b][j] = c; hs[b][j] = h;
            qh[((long)b * Qc + t) * Dc + j] = h;
        }
        __syncthreads();
    }
}

// ======================================================================
// Attention softmax over q + fact weight Wt[b,f] = sum_q p_q * S[b,q,f]
// ======================================================================
__global__ void k_softmax_wt(const float* __restrict__ S, const int* __restrict__ qtext,
                             float* __restrict__ Wt)
{
    int idx = blockIdx.x * blockDim.x + threadIdx.x;
    if (idx >= Bc * Fc) return;
    int b = idx / Fc, f = idx - b * Fc;
    float sv[Qc];
    float mx = -3.4e38f;
#pragma unroll
    for (int q = 0; q < Qc; ++q) {
        float mterm = (qtext[b * Qc + q] != NUM_WORDc) ? 0.0f : VERY_NEGc;
        float s = S[((long)b * Qc + q) * Fc + f] + mterm;
        sv[q] = s;
        mx = fmaxf(mx, s);
    }
    float den = 0.0f, num = 0.0f;
#pragma unroll
    for (int q = 0; q < Qc; ++q) {
        float e = expf(sv[q] - mx);
        den += e;
        num += e * sv[q];
    }
    Wt[idx] = num / den;
}

__global__ void k_bmax(const float* __restrict__ Wt, float* __restrict__ wmax)
{
    __shared__ float red[256];
    int b = blockIdx.x;
    float m = -3.4e38f;
    for (int f = threadIdx.x; f < Fc; f += 256) m = fmaxf(m, Wt[b * Fc + f]);
    red[threadIdx.x] = m;
    __syncthreads();
    for (int s = 128; s; s >>= 1) {
        if (threadIdx.x < s) red[threadIdx.x] = fmaxf(red[threadIdx.x], red[threadIdx.x + s]);
        __syncthreads();
    }
    if (threadIdx.x == 0) wmax[b] = red[0];
}

__global__ void k_wtilde(const float* Wt, const float* wmax, float* out)
{
    int i = blockIdx.x * blockDim.x + threadIdx.x;
    if (i >= Bc * Fc) return;
    out[i] = expf(Wt[i] - wmax[i / Fc]);
}

// ======================================================================
// Sparse scatters
// ======================================================================
__global__ void k_scatter_scalar(const int* __restrict__ ib, const int* __restrict__ io,
                                 const int* __restrict__ is, const float* __restrict__ val,
                                 const float* __restrict__ src, float* __restrict__ out,
                                 int outN, int srcN, int nnz)
{
    int g = blockIdx.x * blockDim.x + threadIdx.x;
    if (g >= nnz) return;
    int b = ib[g];
    atomicAdd(&out[(long)b * outN + io[g]], val[g] * src[(long)b * srcN + is[g]]);
}

__global__ void k_scatter_vec(const int* __restrict__ ib, const int* __restrict__ io,
                              const int* __restrict__ is, const float* __restrict__ val,
                              const float* __restrict__ src, float* __restrict__ out,
                              int outN, int srcN, int nnz)
{
    int g = blockIdx.x * (blockDim.x >> 5) + (threadIdx.x >> 5);  // wave per nnz
    if (g >= nnz) return;
    int lane = threadIdx.x & 31;
    int b = ib[g];
    long o = ((long)b * outN + io[g]) * (long)Dc;
    long s = ((long)b * srcN + is[g]) * (long)Dc;
    float v = val[g];
#pragma unroll
    for (int j = 0; j < 8; ++j) {
        int d = lane + 32 * j;   // coalesced across the wave
        atomicAdd(&out[o + d], v * src[s + d]);
    }
}

// ======================================================================
// Elementwise / small helpers
// ======================================================================
__global__ void k_copy(float* dst, const float* src, long n)
{ long i = (long)blockIdx.x * blockDim.x + threadIdx.x; if (i < n) dst[i] = src[i]; }

__global__ void k_qne_init(const float* __restrict__ qh, float* __restrict__ qne)
{
    int i = blockIdx.x * blockDim.x + threadIdx.x;
    if (i >= Bc * Dc) return;
    int b = i >> 8, d = i & 255;
    qne[i] = qh[((long)b * Qc + (Qc - 1)) * Dc + d];
}

__global__ void k_clampmin(float* x, float v, long n)
{ long i = (long)blockIdx.x * blockDim.x + threadIdx.x; if (i < n) x[i] = fmaxf(x[i], v); }

__global__ void k_ratio(const float* pr, const float* sm, float* r, long n)
{ long i = (long)blockIdx.x * blockDim.x + threadIdx.x; if (i < n) r[i] = pr[i] / sm[i]; }

__global__ void k_mul(float* a, const float* b, long n)
{ long i = (long)blockIdx.x * blockDim.x + threadIdx.x; if (i < n) a[i] *= b[i]; }

__global__ void k_relu_mul_norm(float* x, const float* norm, long n)
{
    long i = (long)blockIdx.x * blockDim.x + threadIdx.x;
    if (i < n) x[i] = fmaxf(x[i], 0.0f) * norm[i >> 8];
}

__global__ void k_relu(float* x, long n)
{ long i = (long)blockIdx.x * blockDim.x + threadIdx.x; if (i < n) x[i] = fmaxf(x[i], 0.0f); }

__global__ void k_prupd(float* pr, const float* tmp, long n)
{
    long i = (long)blockIdx.x * blockDim.x + threadIdx.x;
    if (i < n) pr[i] = PR_LAMBDA * tmp[i] + (1.0f - PR_LAMBDA) * pr[i];
}

__global__ void k_concat(const float* __restrict__ lee, const float* __restrict__ q2e,
                         const float* __restrict__ f2e, float* __restrict__ nxt)
{
    long i = (long)blockIdx.x * blockDim.x + threadIdx.x;
    if (i >= (long)Bc * Ec * Dc) return;
    long row = i >> 8; int d = (int)(i & 255);
    int b = (int)(row / Ec);
    float* o = nxt + row * (3 * Dc);
    o[d]          = lee[i];
    o[Dc + d]     = q2e[b * Dc + d];
    o[2 * Dc + d] = FACT_SCALEc * f2e[i];
}

// small lin for query node (M=8): out[b,n] = qne[b,:] . W[n,:] + bias[n]
__global__ __launch_bounds__(256) void k_lin_small(
    const float* __restrict__ x, const float* __restrict__ W,
    const float* __restrict__ bias, float* __restrict__ out)
{
    __shared__ float xs[Dc];
    int b = blockIdx.x, n = threadIdx.x;
    xs[n] = x[b * Dc + n];
    __syncthreads();
    float acc = bias[n];
    const float* wr = W + (long)n * Dc;
    for (int k = 0; k < Dc; ++k) acc += xs[k] * wr[k];
    out[b * Dc + n] = acc;
}

// qne[b,d] = sum_e pr[b,e] * x[b,e,d]
__global__ __launch_bounds__(256) void k_wsum(const float* __restrict__ pr,
                                              const float* __restrict__ x,
                                              float* __restrict__ qne)
{
    int b = blockIdx.x, d = threadIdx.x;
    float acc = 0.0f;
    for (int e = 0; e < Ec; ++e)
        acc += pr[b * Ec + e] * x[((long)b * Ec + e) * Dc + d];
    qne[b * Dc + d] = acc;
}

// score[r] = lee[r,:] . sW + sb   (wave per row, wave32 shuffle reduce)
__global__ void k_score(const float* __restrict__ lee, const float* __restrict__ sW,
                        const float* __restrict__ sb, float* __restrict__ score, int rows)
{
    int r = blockIdx.x * (blockDim.x >> 5) + (threadIdx.x >> 5);
    if (r >= rows) return;
    int lane = threadIdx.x & 31;
    float acc = 0.0f;
    for (int d = lane; d < Dc; d += 32) acc += lee[(long)r * Dc + d] * sW[d];
    for (int off = 16; off; off >>= 1) acc += __shfl_down(acc, off, 32);
    if (lane == 0) score[r] = acc + sb[0];
}

__global__ void k_loss(const float* __restrict__ score, const float* __restrict__ ans,
                       float* __restrict__ out)
{
    int i = blockIdx.x * blockDim.x + threadIdx.x;
    if (i >= Bc * Ec) return;
    float s = score[i];
    float t = fmaxf(s, 0.0f) - s * ans[i] + log1pf(expf(-fabsf(s)));
    atomicAdd(out, t / (float)(Bc * Ec));
}

// per-batch argmax + pred_dist
__global__ __launch_bounds__(256) void k_pred(const float* __restrict__ score,
                                              const int* __restrict__ local_entity,
                                              float* __restrict__ dout)
{
    __shared__ float bv[256];
    __shared__ int   bi[256];
    int b = blockIdx.x, tid = threadIdx.x;
    float best = -3.4e38f; int bidx = 0;
    for (int e = tid; e < Ec; e += 256) {
        float mask = (local_entity[b * Ec + e] != NUM_ENTITYc) ? 1.0f : 0.0f;
        float sm = score[b * Ec + e] + (1.0f - mask) * VERY_NEGc;
        if (sm > best) { best = sm; bidx = e; }
        dout[1 + Bc + b * Ec + e] = mask / (1.0f + expf(-sm));   // pred_dist
    }
    bv[tid] = best; bi[tid] = bidx;
    __syncthreads();
    for (int s = 128; s; s >>= 1) {
        if (tid < s) {
            if (bv[tid + s] > bv[tid] || (bv[tid + s] == bv[tid] && bi[tid + s] < bi[tid])) {
                bv[tid] = bv[tid + s]; bi[tid] = bi[tid + s];
            }
        }
        __syncthreads();
    }
    if (tid == 0) dout[1 + b] = (float)bi[0];
}

// ======================================================================
// Host orchestration
// ======================================================================
extern "C" void kernel_launch(void* const* d_in, const int* in_sizes, int n_in,
                              void* d_out, int out_size, void* d_ws, size_t ws_size,
                              hipStream_t stream)
{
    (void)in_sizes; (void)n_in; (void)out_size; (void)ws_size;

    const int*   local_entity  = (const int*)  d_in[0];
    const float* q2e_adj       = (const float*)d_in[1];
    const int*   kb_fact_rel   = (const int*)  d_in[2];
    const int*   query_text    = (const int*)  d_in[3];
    const float* answer_dist   = (const float*)d_in[4];
    const int*   e2f           = (const int*)  d_in[5];
    const float* e2f_val       = (const float*)d_in[6];
    const int*   f2e           = (const int*)  d_in[7];
    const float* f2e_val       = (const float*)d_in[8];
    const float* entity_table  = (const float*)d_in[9];
    const float* ent_lin_W     = (const float*)d_in[10];
    const float* ent_lin_b     = (const float*)d_in[11];
    const float* relation_tab  = (const float*)d_in[12];
    const float* rel_lin_W     = (const float*)d_in[13];
    const float* rel_lin_b     = (const float*)d_in[14];
    const float* word_table    = (const float*)d_in[15];
    const float* lstm_Wih      = (const float*)d_in[16];
    const float* lstm_Whh      = (const float*)d_in[17];
    const float* lstm_bih      = (const float*)d_in[18];
    const float* lstm_bhh      = (const float*)d_in[19];
    const float* q2e_W         = (const float*)d_in[20];
    const float* q2e_b         = (const float*)d_in[21];
    const float* e2q_W         = (const float*)d_in[22];
    const float* e2q_b         = (const float*)d_in[23];
    const float* e2e_W         = (const float*)d_in[24];
    const float* e2e_b         = (const float*)d_in[25];
    const float* head_W        = (const float*)d_in[26];
    const float* head_b        = (const float*)d_in[27];
    const float* tail_W        = (const float*)d_in[28];
    const float* tail_b        = (const float*)d_in[29];
    const float* self_W        = (const float*)d_in[30];
    const float* self_b        = (const float*)d_in[31];
    const float* score_W       = (const float*)d_in[32];
    const float* score_b       = (const float*)d_in[33];
    float* out = (float*)d_out;

    // ---- workspace carve-up (floats) ----
    float* wsf = (float*)d_ws;
    size_t off = 0;
    auto alloc = [&](size_t n) { float* p = wsf + off; off += (n + 63) & ~(size_t)63; return p; };
    float* qh     = alloc((size_t)Bc * Qc * Dc);
    float* qne    = alloc((size_t)Bc * Dc);
    float* lfe    = alloc((size_t)Bc * Fc * Dc);
    float* S      = alloc((size_t)Bc * Qc * Fc);
    float* Wt     = alloc((size_t)Bc * Fc);
    float* Wtil   = alloc((size_t)Bc * Fc);
    float* wmax   = alloc(Bc);
    float* e2fsm  = alloc((size_t)Bc * Ec);
    float* pr     = alloc((size_t)Bc * Ec);
    float* prtmp  = alloc((size_t)Bc * Ec);
    float* ratio  = alloc((size_t)Bc * Ec);
    float* facc   = alloc((size_t)Bc * Fc);      // becomes e2f_norm
    float* q2eo   = alloc((size_t)Bc * Dc);
    float* lee    = alloc((size_t)Bc * Ec * Dc);
    float* eBuf1  = alloc((size_t)Bc * Ec * Dc);
    float* eBuf2  = alloc((size_t)Bc * Ec * Dc);
    float* eBuf3  = alloc((size_t)Bc * Ec * Dc);
    float* bigF2  = alloc((size_t)Bc * Fc * Dc);
    float* bigF3  = alloc((size_t)Bc * Fc * Dc);
    float* nxt    = alloc((size_t)Bc * Ec * 3 * Dc);
    float* scoreb = alloc((size_t)Bc * Ec);

    auto gemm = [&](const float* A, int lda, const int* gidx, const float* W, int ldw,
                    const float* bias, float* C, int ldc, int M, int N, int K,
                    float alpha, int relu) {
        dim3 grid((M + BM - 1) / BM, (N + BNT - 1) / BNT);
        if (gidx)
            k_gemm_bf16<true ><<<grid, dim3(256), 0, stream>>>(A, lda, gidx, W, ldw, bias,
                                                               C, ldc, M, N, K, alpha, relu);
        else
            k_gemm_bf16<false><<<grid, dim3(256), 0, stream>>>(A, lda, gidx, W, ldw, bias,
                                                               C, ldc, M, N, K, alpha, relu);
    };
    auto elems = [&](long n) { return dim3((unsigned)((n + 255) / 256)); };

    // 1) LSTM encoder -> qh, query node init
    k_lstm<<<1, 1024, 0, stream>>>(query_text, word_table, lstm_Wih, lstm_Whh,
                                   lstm_bih, lstm_bhh, qh);
    k_qne_init<<<elems(Bc * Dc), 256, 0, stream>>>(qh, qne);

    // 2) fact embeddings: lfe = relation_table[kb_fact_rel] @ rel_lin_W^T + b
    gemm(relation_tab, Dc, kb_fact_rel, rel_lin_W, Dc, rel_lin_b, lfe, Dc,
         Bc * Fc, Dc, Dc, 1.0f, 0);

    // 3) attention scores S[b,q,f] = qh . lfe / sqrt(D)   (batched WMMA GEMM)
    for (int b = 0; b < Bc; ++b)
        gemm(qh + (long)b * Qc * Dc, Dc, nullptr, lfe + (long)b * Fc * Dc, Dc,
             nullptr, S + (long)b * Qc * Fc, Fc, Qc, Fc, Dc, 1.0f / 16.0f, 0);

    // 4) softmax over q + Wt, W_tilde
    k_softmax_wt<<<elems(Bc * Fc), 256, 0, stream>>>(S, query_text, Wt);
    k_bmax<<<Bc, 256, 0, stream>>>(Wt, wmax);
    k_wtilde<<<elems(Bc * Fc), 256, 0, stream>>>(Wt, wmax, Wtil);

    // 5) e2f_softmax[b,e] = max(sum e2f Wtil[b,f], eps)
    hipMemsetAsync(e2fsm, 0, (size_t)Bc * Ec * sizeof(float), stream);
    k_scatter_scalar<<<elems(NNZc), 256, 0, stream>>>(
        e2f, e2f + 2 * NNZc, e2f + NNZc, e2f_val, Wtil, e2fsm, Ec, Fc, NNZc);
    k_clampmin<<<elems(Bc * Ec), 256, 0, stream>>>(e2fsm, VERY_SMALLc, (long)Bc * Ec);

    // 6) pagerank init, entity embeddings
    k_copy<<<elems(Bc * Ec), 256, 0, stream>>>(pr, q2e_adj, (long)Bc * Ec);
    gemm(entity_table, Dc, local_entity, ent_lin_W, Dc, ent_lin_b, lee, Dc,
         Bc * Ec, Dc, Dc, 1.0f, 0);

    // 7) graph propagation layers
    const long nED = (long)Bc * Ec * Dc;
    const long nFD = (long)Bc * Fc * Dc;
    for (int i = 0; i < Lc; ++i) {
        const float* q2eWi = q2e_W  + (size_t)i * Dc * Dc;
        const float* headWi= head_W + (size_t)i * Dc * Dc;
        const float* tailWi= tail_W + (size_t)i * Dc * Dc;
        const float* selfWi= self_W + (size_t)i * Dc * Dc;
        const float* e2qWi = e2q_W  + (size_t)i * Dc * 3 * Dc;
        const float* e2eWi = e2e_W  + (size_t)i * Dc * 3 * Dc;

        // query->entity message (B x D, tiny)
        k_lin_small<<<Bc, Dc, 0, stream>>>(qne, q2eWi, q2e_b + i * Dc, q2eo);

        // head lin on entities, self lin on facts, scatter e2f
        gemm(lee, Dc, nullptr, headWi, Dc, head_b + i * Dc, eBuf1, Dc, Bc * Ec, Dc, Dc, 1.0f, 0);
        gemm(lfe, Dc, nullptr, selfWi, Dc, self_b + i * Dc, bigF2, Dc, Bc * Fc, Dc, Dc, 1.0f, 0);
        k_scatter_vec<<<(NNZc + 7) / 8, 256, 0, stream>>>(
            e2f, e2f + NNZc, e2f + 2 * NNZc, e2f_val, eBuf1, bigF2, Fc, Ec, NNZc);

        // e2f_norm = Wtil * scatter(pagerank / e2f_softmax)
        k_ratio<<<elems(Bc * Ec), 256, 0, stream>>>(pr, e2fsm, ratio, (long)Bc * Ec);
        hipMemsetAsync(facc, 0, (size_t)Bc * Fc * sizeof(float), stream);
        k_scatter_scalar<<<elems(NNZc), 256, 0, stream>>>(
            e2f, e2f + NNZc, e2f + 2 * NNZc, e2f_val, ratio, facc, Fc, Ec, NNZc);
        k_mul<<<elems(Bc * Fc), 256, 0, stream>>>(facc, Wtil, (long)Bc * Fc);   // facc = norm

        // e2f_emb = relu(bigF2) * norm
        k_relu_mul_norm<<<elems(nFD), 256, 0, stream>>>(bigF2, facc, nFD);

        // tail lin, self lin on entities, scatter f2e, relu
        gemm(bigF2, Dc, nullptr, tailWi, Dc, tail_b + i * Dc, bigF3, Dc, Bc * Fc, Dc, Dc, 1.0f, 0);
        gemm(lee, Dc, nullptr, selfWi, Dc, self_b + i * Dc, eBuf1, Dc, Bc * Ec, Dc, Dc, 1.0f, 0);
        k_scatter_vec<<<(NNZc + 7) / 8, 256, 0, stream>>>(
            f2e, f2e + NNZc, f2e + 2 * NNZc, f2e_val, bigF3, eBuf1, Ec, Fc, NNZc);
        k_relu<<<elems(nED), 256, 0, stream>>>(eBuf1, nED);                     // f2e_emb

        // pagerank update
        hipMemsetAsync(prtmp, 0, (size_t)Bc * Ec * sizeof(float), stream);
        k_scatter_scalar<<<elems(NNZc), 256, 0, stream>>>(
            f2e, f2e + NNZc, f2e + 2 * NNZc, f2e_val, facc, prtmp, Ec, Fc, NNZc);
        k_prupd<<<elems(Bc * Ec), 256, 0, stream>>>(pr, prtmp, (long)Bc * Ec);

        // nxt = [lee | q2eo bcast | 3*f2e_emb], then e2q / e2e
        k_concat<<<elems(nED), 256, 0, stream>>>(lee, q2eo, eBuf1, nxt);
        gemm(nxt, 3 * Dc, nullptr, e2qWi, 3 * Dc, e2q_b + i * Dc, eBuf2, Dc,
             Bc * Ec, Dc, 3 * Dc, 1.0f, 0);
        k_wsum<<<Bc, Dc, 0, stream>>>(pr, eBuf2, qne);
        gemm(nxt, 3 * Dc, nullptr, e2eWi, 3 * Dc, e2e_b + i * Dc, eBuf3, Dc,
             Bc * Ec, Dc, 3 * Dc, 1.0f, 1 /*relu*/);

        float* t = lee; lee = eBuf3; eBuf3 = t;   // new entity embeddings
    }

    // 8) score, loss, pred, pred_dist
    k_score<<<(Bc * Ec + 7) / 8, 256, 0, stream>>>(lee, score_W, score_b, scoreb, Bc * Ec);
    hipMemsetAsync(out, 0, sizeof(float), stream);
    k_loss<<<elems(Bc * Ec), 256, 0, stream>>>(scoreb, answer_dist, out);
    k_pred<<<Bc, 256, 0, stream>>>(scoreb, local_entity, out);
}
